// Encoder_50448685858991
// MI455X (gfx1250) — matrix operand order repaired and checked by reading, not compiled, not run on hardware
//
#include <hip/hip_runtime.h>
#include <hip/hip_bf16.h>

// ---------------------------------------------------------------------------
// Problem constants (fixed by the reference)
// ---------------------------------------------------------------------------
#define BB   2
#define SS   2048
#define DD   1024
#define HH   16
#define DHD  64
#define HID_ 4096

typedef __attribute__((ext_vector_type(16))) __bf16 v16bf;
typedef __attribute__((ext_vector_type(8)))  __bf16 v8bf;
typedef __attribute__((ext_vector_type(8)))  float  v8f;

// Load 8 consecutive bf16 (16 bytes) — lowers to global_load_b128 / ds_load_b128
__device__ __forceinline__ v8bf ld8(const __bf16* p) {
    return *(const v8bf*)p;
}
__device__ __forceinline__ v16bf cat16(v8bf lo, v8bf hi) {
    v16bf r;
#pragma unroll
    for (int i = 0; i < 8; ++i) { r[i] = lo[i]; r[8 + i] = hi[i]; }
    return r;
}
__device__ __forceinline__ v8f wmma_bf16(v16bf a, v16bf b, v8f c) {
    return __builtin_amdgcn_wmma_f32_16x16x32_bf16(
        /*neg_a=*/false, a, /*neg_b=*/false, b,
        /*c_mod=*/(short)0, c, /*reuse_a=*/false, /*reuse_b=*/false);
}

// ---------------------------------------------------------------------------
// K0: f32 -> bf16 convert (grid-stride)
// ---------------------------------------------------------------------------
__global__ void cvt_kernel(const float* __restrict__ src, __bf16* __restrict__ dst, long long n) {
    long long i = (long long)blockIdx.x * blockDim.x + threadIdx.x;
    long long stride = (long long)gridDim.x * blockDim.x;
    for (; i < n; i += stride) dst[i] = (__bf16)src[i];
}

// ---------------------------------------------------------------------------
// K1: batched transpose + convert: dst[b][c][r] = (bf16) src[b][r][c]
// ---------------------------------------------------------------------------
__global__ void tconv_kernel(const float* __restrict__ src, __bf16* __restrict__ dst,
                             int rows, int cols) {
    long long nb = (long long)rows * cols;
    const float* s = src + (long long)blockIdx.y * nb;
    __bf16*      d = dst + (long long)blockIdx.y * nb;
    long long i = (long long)blockIdx.x * blockDim.x + threadIdx.x;
    long long stride = (long long)gridDim.x * blockDim.x;
    for (; i < nb; i += stride) {
        long long r = i / cols, c = i % cols;
        d[c * (long long)rows + r] = (__bf16)s[i];
    }
}

// ---------------------------------------------------------------------------
// K2: batched bf16 WMMA GEMM:  C[M,N] = act( A[M,K] * Bt[N,K]^T + bias )
//   A row-major along K, Bt row-major along K (B pre-transposed), so every
//   WMMA fragment is two contiguous b128 loads per lane (ISA 7.12.2 layouts).
//
//   Register blocking: each wave owns a 64x64 C tile = 4x4 subtiles of
//   16x16 (16 v8f accumulators, 128 VGPRs). Per 32-wide K step it loads
//   4 A-fragments + 4 B-fragments (8 KB/wave) and issues 16
//   v_wmma_f32_16x16x32_bf16 -> 512 B per WMMA (32 FLOP/byte of L0 traffic).
//
//   Block = 256 threads = 8 waves stacked in M: workgroup tile 512 x 64.
//   Outputs optionally: f32 C, bf16 C, bf16 C^T (attention needs q and q^T).
// ---------------------------------------------------------------------------
__global__ __launch_bounds__(256) void gemm_kernel(
    const __bf16* __restrict__ A, const __bf16* __restrict__ Bt,
    const float* __restrict__ bias,
    float* __restrict__ Cf, __bf16* __restrict__ Cb, __bf16* __restrict__ CbT,
    int M, int N, int K,
    long long sA, long long sB, long long sBias,
    long long sCf, long long sCb, long long sCbT,
    int aDiv, int bMod, int relu)
{
    const int z    = blockIdx.z;
    const __bf16* Ab = A  + (long long)(z / aDiv) * sA;
    const __bf16* Bb = Bt + (long long)(z % bMod) * sB;

    const int tid  = threadIdx.x;
    const int w    = tid >> 5;
    const int lane = tid & 31;
    const int half = lane >> 4;
    const int lm   = lane & 15;

    const int m0 = blockIdx.x * 512 + w * 64;
    const int n0 = blockIdx.y * 64;

    const __bf16* rowA[4];
    const __bf16* rowB[4];
#pragma unroll
    for (int i = 0; i < 4; ++i) {
        rowA[i] = Ab + (long long)(m0 + i * 16 + lm) * K;
        rowB[i] = Bb + (long long)(n0 + i * 16 + lm) * K;
    }

    v8f acc[4][4] = {};
    for (int k0 = 0; k0 < K; k0 += 32) {
        if (k0 + 32 < K) {
            __builtin_prefetch(rowA[0] + k0 + 32, 0, 1);   // global_prefetch_b8
            __builtin_prefetch(rowB[0] + k0 + 32, 0, 1);
        }
        v16bf am[4], bn[4];
#pragma unroll
        for (int i = 0; i < 4; ++i)
            am[i] = cat16(ld8(rowA[i] + k0 + half * 8),
                          ld8(rowA[i] + k0 + 16 + half * 8));
#pragma unroll
        for (int i = 0; i < 4; ++i)
            bn[i] = cat16(ld8(rowB[i] + k0 + half * 8),
                          ld8(rowB[i] + k0 + 16 + half * 8));
#pragma unroll
        for (int ms = 0; ms < 4; ++ms)
#pragma unroll
            for (int nt = 0; nt < 4; ++nt)
                acc[ms][nt] = wmma_bf16(am[ms], bn[nt], acc[ms][nt]);
    }

    // Epilogue: C layout — lane: n = n0+nt*16+lm ; r: m = m0+ms*16+half*8+r
#pragma unroll
    for (int ms = 0; ms < 4; ++ms) {
#pragma unroll
        for (int nt = 0; nt < 4; ++nt) {
            const int n = n0 + nt * 16 + lm;
            float bv = bias ? bias[(long long)(z % bMod) * sBias + n] : 0.0f;
#pragma unroll
            for (int r = 0; r < 8; ++r) {
                const int m = m0 + ms * 16 + half * 8 + r;
                float v = acc[ms][nt][r] + bv;
                if (relu) v = fmaxf(v, 0.0f);
                if (Cf)  Cf [(long long)z * sCf  + (long long)m * N + n] = v;
                if (Cb)  Cb [(long long)z * sCb  + (long long)m * N + n] = (__bf16)v;
                if (CbT) CbT[(long long)z * sCbT + (long long)n * M + m] = (__bf16)v;
            }
        }
    }
}

// ---------------------------------------------------------------------------
// K3: fused self-attention, single-pass flash, one (b, h, 16-row block)
//   per workgroup. q == k == v (reference quirk).
//
//   128 threads = 4 waves; wave w owns a 512-column key chunk. Per 32-column
//   block: 4 WMMAs for the QK^T score tiles, shuffle-reduced running
//   (max, sumexp) with accumulator rescale by exp(m_old - m_new)
//   (uniform over each 16-lane group -> 32 v_mul_f32), unnormalized
//   P -> bf16 through a per-wave LDS transpose stage into the WMMA A layout,
//   then 4 WMMAs of P*V against q^T rows. Per-wave partials are combined
//   split-K style: (m_w, s_w) -> LDS, global (M, S) formed once, each wave
//   folds exp(m_w - M)/S into its accumulator before the ds_add_f32 combine.
// ---------------------------------------------------------------------------
__global__ __launch_bounds__(128) void attn_kernel(
    const __bf16* __restrict__ qb,   // [B,H,S,DH] bf16
    const __bf16* __restrict__ qbT,  // [B,H,DH,S] bf16
    float* __restrict__ attn)        // [B,S,D] f32
{
    const int mt = blockIdx.x, h = blockIdx.y, b = blockIdx.z;
    const int tid  = threadIdx.x;
    const int w    = tid >> 5;
    const int lane = tid & 31;
    const int half = lane >> 4;
    const int lm   = lane & 15;
    const int mbase = mt * 16;

    const __bf16* q  = qb  + ((long long)(b * HH + h)) * SS * DHD;
    const __bf16* qT = qbT + ((long long)(b * HH + h)) * DHD * SS;

    __shared__ __align__(16) __bf16 pstage[4][16][32];
    __shared__ float wmax[4][16], wsum[4][16];
    __shared__ float fmaxs[16], finv[16];
    __shared__ float outacc[16][64];

    // Zero the cross-wave output accumulator (atomics happen after barriers)
    for (int i = tid; i < 16 * 64; i += 128) ((float*)outacc)[i] = 0.0f;

    // Q A-fragments for this row block (reused by every tile): K = DH = 64
    const __bf16* rq = q + (long long)(mbase + lm) * DHD;
    const v16bf qa0 = cat16(ld8(rq + half * 8),      ld8(rq + 16 + half * 8));
    const v16bf qa1 = cat16(ld8(rq + 32 + half * 8), ld8(rq + 48 + half * 8));

    // Running softmax state for rows half*8 + r (uniform across the 16-lane
    // group), and the unnormalized P*V accumulator.
    float rm[8], rs[8];
#pragma unroll
    for (int r = 0; r < 8; ++r) { rm[r] = -1e30f; rs[r] = 0.0f; }
    v8f oacc[4] = {};

    for (int blk = 0; blk < 16; ++blk) {
        const int col0 = w * 512 + blk * 32;

        // ---- scores for 32 columns: two 16-wide tiles ----
        v8f sc[2];
#pragma unroll
        for (int t = 0; t < 2; ++t) {
            const __bf16* rb = q + (long long)(col0 + t * 16 + lm) * DHD;
            v16bf b0 = cat16(ld8(rb + half * 8),      ld8(rb + 16 + half * 8));
            v16bf b1 = cat16(ld8(rb + 32 + half * 8), ld8(rb + 48 + half * 8));
            v8f a = {};
            a = wmma_bf16(qa0, b0, a);
            sc[t] = wmma_bf16(qa1, b1, a);
        }

        // ---- online softmax update ----
        float scale[8];
#pragma unroll
        for (int r = 0; r < 8; ++r) {
            const float v0 = sc[0][r] * 0.125f;              // 1/sqrt(DH)
            const float v1 = sc[1][r] * 0.125f;
            float tmax = fmaxf(v0, v1);
#pragma unroll
            for (int m = 1; m < 16; m <<= 1)                 // reduce over lm 0..15
                tmax = fmaxf(tmax, __shfl_xor(tmax, m, 32));
            const float nm = fmaxf(rm[r], tmax);
            scale[r] = __expf(rm[r] - nm);
            rm[r] = nm;
            const float e0 = __expf(v0 - nm);
            const float e1 = __expf(v1 - nm);
            float te = e0 + e1;
#pragma unroll
            for (int m = 1; m < 16; m <<= 1)
                te += __shfl_xor(te, m, 32);
            rs[r] = rs[r] * scale[r] + te;
            // unnormalized P in (m, k) layout for the A-fragment reload
            pstage[w][half * 8 + r][lm]      = (__bf16)e0;
            pstage[w][half * 8 + r][16 + lm] = (__bf16)e1;
        }
        // rescale the running P*V accumulator to the new max
#pragma unroll
        for (int nt = 0; nt < 4; ++nt)
#pragma unroll
            for (int r = 0; r < 8; ++r)
                oacc[nt][r] *= scale[r];

        __syncthreads();
        // P as A-fragment: lane holds row lm
        v16bf pa = cat16(ld8(&pstage[w][lm][half * 8]),
                         ld8(&pstage[w][lm][16 + half * 8]));
#pragma unroll
        for (int nt = 0; nt < 4; ++nt) {
            const __bf16* rt = qT + (long long)(nt * 16 + lm) * SS + col0 + half * 8;
            v16bf bt = cat16(ld8(rt), ld8(rt + 16));
            oacc[nt] = wmma_bf16(pa, bt, oacc[nt]);
        }
        __syncthreads();
    }

    // ---- split-K combine of per-wave partials ----
    if (lm == 0) {
#pragma unroll
        for (int r = 0; r < 8; ++r) {
            wmax[w][half * 8 + r] = rm[r];
            wsum[w][half * 8 + r] = rs[r];
        }
    }
    __syncthreads();
    if (tid < 16) {
        float M = -1e30f;
        for (int w2 = 0; w2 < 4; ++w2) M = fmaxf(M, wmax[w2][tid]);
        float Ssum = 0.0f;
        for (int w2 = 0; w2 < 4; ++w2) Ssum += wsum[w2][tid] * __expf(wmax[w2][tid] - M);
        fmaxs[tid] = M;
        finv[tid]  = 1.0f / Ssum;
    }
    __syncthreads();

    float fac[8];
#pragma unroll
    for (int r = 0; r < 8; ++r)
        fac[r] = __expf(rm[r] - fmaxs[half * 8 + r]) * finv[half * 8 + r];

#pragma unroll
    for (int nt = 0; nt < 4; ++nt)
#pragma unroll
        for (int r = 0; r < 8; ++r)
            atomicAdd(&outacc[half * 8 + r][nt * 16 + lm], oacc[nt][r] * fac[r]);
    __syncthreads();

    for (int i = tid; i < 1024; i += 128) {
        const int m = i >> 6, n = i & 63;
        attn[((long long)b * SS + mbase + m) * DD + h * DHD + n] = outacc[m][n];
    }
}

// ---------------------------------------------------------------------------
// K4: residual add + LayerNorm over D=1024; one block (256 thr) per row.
//     Writes f32 always, bf16 optionally (next GEMM input).
// ---------------------------------------------------------------------------
__global__ __launch_bounds__(256) void ln_kernel(
    const float* __restrict__ xa, const float* __restrict__ xr,
    const float* __restrict__ g, const float* __restrict__ be,
    float* __restrict__ of, __bf16* __restrict__ ob)
{
    const long long row = blockIdx.x;
    const int tid = threadIdx.x;
    const float* pa = xa + row * DD;
    const float* pr = xr + row * DD;

    float v[4], s1 = 0.0f, s2 = 0.0f;
#pragma unroll
    for (int i = 0; i < 4; ++i) {
        const int c = i * 256 + tid;
        v[i] = pa[c] + pr[c];
        s1 += v[i];
        s2 += v[i] * v[i];
    }
    __shared__ float sh1[256], sh2[256];
    sh1[tid] = s1; sh2[tid] = s2;
    __syncthreads();
    for (int off = 128; off > 0; off >>= 1) {
        if (tid < off) { sh1[tid] += sh1[tid + off]; sh2[tid] += sh2[tid + off]; }
        __syncthreads();
    }
    const float mean = sh1[0] * (1.0f / DD);
    const float var  = sh2[0] * (1.0f / DD) - mean * mean;
    const float rsd  = rsqrtf(var + 1e-5f);
#pragma unroll
    for (int i = 0; i < 4; ++i) {
        const int c = i * 256 + tid;
        const float y = (v[i] - mean) * rsd * g[c] + be[c];
        of[row * DD + c] = y;
        if (ob) ob[row * DD + c] = (__bf16)y;
    }
}

// ---------------------------------------------------------------------------
// Host-side orchestration
// ---------------------------------------------------------------------------
extern "C" void kernel_launch(void* const* d_in, const int* in_sizes, int n_in,
                              void* d_out, int out_size, void* d_ws, size_t ws_size,
                              hipStream_t stream) {
    (void)in_sizes; (void)n_in; (void)out_size; (void)ws_size;

    const float* x    = (const float*)d_in[0];
    const float* Wq   = (const float*)d_in[1];
    const float* bq   = (const float*)d_in[2];
    const float* ln1g = (const float*)d_in[3];
    const float* ln1b = (const float*)d_in[4];
    const float* W1   = (const float*)d_in[5];
    const float* b1   = (const float*)d_in[6];
    const float* W2   = (const float*)d_in[7];
    const float* b2   = (const float*)d_in[8];
    const float* ln2g = (const float*)d_in[9];
    const float* ln2b = (const float*)d_in[10];
    float* out = (float*)d_out;

    const long long nX = (long long)BB * SS * DD;      // 4,194,304
    const long long nH = (long long)BB * SS * HID_;    // 16,777,216

    // Workspace carve (256B-aligned)
    char* ws = (char*)d_ws;
    size_t off = 0;
    auto carve = [&](size_t bytes) -> char* {
        char* p = ws + off;
        off += (bytes + 255) & ~(size_t)255;
        return p;
    };
    __bf16* xb   = (__bf16*)carve(nX * 2);                         // x bf16
    __bf16* WqT  = (__bf16*)carve((size_t)HH * DHD * DD * 2);      // [H,DH,D]
    __bf16* W1T  = (__bf16*)carve((size_t)DD * HID_ * 2);          // [HID,D]
    __bf16* W2T  = (__bf16*)carve((size_t)HID_ * DD * 2);          // [D,HID]
    __bf16* qb   = (__bf16*)carve(nX * 2);                         // [B,H,S,DH]
    __bf16* qbT  = (__bf16*)carve(nX * 2);                         // [B,H,DH,S]
    float*  attn = (float*)carve(nX * 4);                          // [B,S,D]
    float*  x1   = (float*)carve(nX * 4);                          // [B,S,D]
    __bf16* x1b  = (__bf16*)carve(nX * 2);
    __bf16* hb   = (__bf16*)carve(nH * 2);                         // [B*S,HID]
    float*  ff   = (float*)carve(nX * 4);

    // K0/K1: precision convert + weight transposes (cheap, bandwidth-bound)
    cvt_kernel<<<4096, 256, 0, stream>>>(x, xb, nX);
    tconv_kernel<<<dim3(256, HH), 256, 0, stream>>>(Wq, WqT, DD, DHD);
    tconv_kernel<<<dim3(8192, 1), 256, 0, stream>>>(W1, W1T, DD, HID_);
    tconv_kernel<<<dim3(8192, 1), 256, 0, stream>>>(W2, W2T, HID_, DD);

    // K2a: Q projection (batched over B*H), writes q and q^T in bf16
    gemm_kernel<<<dim3(SS / 512, DHD / 64, BB * HH), 256, 0, stream>>>(
        xb, WqT, bq,
        /*Cf=*/nullptr, /*Cb=*/qb, /*CbT=*/qbT,
        /*M=*/SS, /*N=*/DHD, /*K=*/DD,
        /*sA=*/(long long)SS * DD, /*sB=*/(long long)DHD * DD, /*sBias=*/DHD,
        /*sCf=*/0, /*sCb=*/(long long)SS * DHD, /*sCbT=*/(long long)DHD * SS,
        /*aDiv=*/HH, /*bMod=*/HH, /*relu=*/0);

    // K3: fused attention (softmax(QQ^T/8) Q), one block per (row16, h, b)
    attn_kernel<<<dim3(SS / 16, HH, BB), 128, 0, stream>>>(qb, qbT, attn);

    // K4: x1 = LN(x + attn)
    ln_kernel<<<dim3((unsigned)(BB * SS)), 256, 0, stream>>>(x, attn, ln1g, ln1b, x1, x1b);

    // K2b: FFN1 with fused ReLU -> hb bf16
    gemm_kernel<<<dim3((BB * SS) / 512, HID_ / 64, 1), 256, 0, stream>>>(
        x1b, W1T, b1,
        nullptr, hb, nullptr,
        BB * SS, HID_, DD,
        0, 0, 0, 0, (long long)0, 0,
        1, 1, /*relu=*/1);

    // K2c: FFN2 -> ff f32
    gemm_kernel<<<dim3((BB * SS) / 512, DD / 64, 1), 256, 0, stream>>>(
        hb, W2T, b2,
        ff, nullptr, nullptr,
        BB * SS, DD, HID_,
        0, 0, 0, 0, 0, 0,
        1, 1, /*relu=*/0);

    // K5: out = LN(x1 + ff)
    ln_kernel<<<dim3((unsigned)(BB * SS)), 256, 0, stream>>>(x1, ff, ln2g, ln2b, out, nullptr);
}
